// Chebs_VGAE_51221779972528
// MI455X (gfx1250) — matrix operand order repaired and checked by reading, not compiled, not run on hardware
//
#include <hip/hip_runtime.h>

// ---------------------------------------------------------------------------
// ChebConv VGAE encoder for MI455X (gfx1250, wave32, WMMA).
//   N=50000 nodes, E=800000 edges, IN=HID=128, OUT=64, K1=16, K2=8.
// Dense Tx_k @ W[k] GEMMs use v_wmma_f32_16x16x32_bf16 (f32 accumulate) with
// weights pre-transposed/converted to bf16 so each B fragment is ONE
// contiguous 32-byte load (no strided b32 loads, no 64-bit address chains).
// Chebyshev recurrence is fused into the scatter: nxt = -prev, then
// nxt += 2*w * gather  (saves an elementwise combine pass and a buffer).
// Layers 2&3 share the Chebyshev basis; biases before BN cancel and are skipped.
// ---------------------------------------------------------------------------

#define N_NODES 50000
#define E_EDGES 800000
#define C_HID   128
#define C_OUT   64
#define K1      16
#define K2      8

typedef __attribute__((ext_vector_type(16))) __bf16 v16bf;
typedef __attribute__((ext_vector_type(8)))  float  v8f;

// ---------------- utility kernels ----------------

__global__ void k_zero(float* __restrict__ p, long n) {
  long i = (long)blockIdx.x * blockDim.x + threadIdx.x;
  long s = (long)gridDim.x * blockDim.x;
  for (; i < n; i += s) p[i] = 0.0f;
}

__global__ void k_deg(const int* __restrict__ src, float* __restrict__ deg, int e_cnt) {
  int e = blockIdx.x * blockDim.x + threadIdx.x;
  if (e < e_cnt) atomicAdd(&deg[src[e]], 1.0f);
}

__global__ void k_dinv(const float* __restrict__ deg, float* __restrict__ dinv, int n) {
  int i = blockIdx.x * blockDim.x + threadIdx.x;
  if (i < n) {
    float d = deg[i];
    dinv[i] = (d > 0.0f) ? rsqrtf(fmaxf(d, 1.0f)) : 0.0f;
  }
}

__global__ void k_edgew(const int* __restrict__ src, const int* __restrict__ dst,
                        const float* __restrict__ dinv, float* __restrict__ w, int e_cnt) {
  int e = blockIdx.x * blockDim.x + threadIdx.x;
  if (e < e_cnt) w[e] = -dinv[src[e]] * dinv[dst[e]];
}

// dst_buf = -src_buf  (may alias; used to seed Tx_k = -Tx_{k-2} before scatter)
__global__ void k_initneg(float4* __restrict__ o, const float4* __restrict__ a, long n4) {
  long i = (long)blockIdx.x * blockDim.x + threadIdx.x;
  long s = (long)gridDim.x * blockDim.x;
  for (; i < n4; i += s) {
    float4 v = a[i];
    float4 r;
    r.x = -v.x; r.y = -v.y; r.z = -v.z; r.w = -v.w;
    o[i] = r;
  }
}

// P[dst, :] += scale * w[e] * Z[src, :]  (128 wide, float4 chunk per lane)
__global__ void k_spmm(const float* __restrict__ Z, float* __restrict__ P,
                       const int* __restrict__ src, const int* __restrict__ dst,
                       const float* __restrict__ w, float scale, int e_cnt) {
  long idx = (long)blockIdx.x * blockDim.x + threadIdx.x;
  int e = (int)(idx >> 5);
  if (e >= e_cnt) return;
  int q = (int)(idx & 31);
  int s = src[e], d = dst[e];
  float wv = scale * w[e];
  float4 z = *(const float4*)(Z + (size_t)s * C_HID + q * 4);
  float* p = P + (size_t)d * C_HID + q * 4;
  atomicAdd(p + 0, wv * z.x);
  atomicAdd(p + 1, wv * z.y);
  atomicAdd(p + 2, wv * z.z);
  atomicAdd(p + 3, wv * z.w);
}

// Transpose+convert weights: W[k][kd][CD] (f32) -> Wt[k][n][kd=0..127] (bf16)
__global__ void k_wt(const float* __restrict__ W, __bf16* __restrict__ Wt,
                     int K, int CD) {
  long total = (long)K * CD * C_HID;
  long i = (long)blockIdx.x * blockDim.x + threadIdx.x;
  long s = (long)gridDim.x * blockDim.x;
  for (; i < total; i += s) {
    int kd = (int)(i & (C_HID - 1));
    long r = i >> 7;
    int n = (int)(r % CD);
    int k = (int)(r / CD);
    Wt[i] = (__bf16)W[((long)k * C_HID + kd) * CD + n];
  }
}

// ---------------- WMMA GEMM-accumulate ----------------
// O[N x CD] += A[N x 128] @ W[128 x CD], W pre-transposed as Wt[CD][128] bf16.
// One wave -> one 16x16 tile; 4 k-steps of v_wmma_f32_16x16x32_bf16.
// A frag (16-bit A 16x32, wave32): M = lane&15,
//   elem i: K = kk*32 + hi*16 + i
// B frag (16-bit B 32x16, wave32): N = lane&15, elem i: K = kk*32 + hi*16 + i
//   -> contiguous v16bf load from Wt[n0*128 + kk*32 + hi*16]
// C/D (f32 16x16): vgpr r -> M = r + hi*8, N = lane&15
__global__ void __launch_bounds__(256)
k_gemm_bf16(const float* __restrict__ A, const __bf16* __restrict__ Wt,
            float* __restrict__ O, int CD, int tiles_n, int total_tiles) {
  int wid = (int)((blockIdx.x * blockDim.x + threadIdx.x) >> 5);  // wave-uniform
  if (wid >= total_tiles) return;                                 // whole wave exits together
  int lane = (int)(threadIdx.x & 31);
  int ln = lane & 15;
  int hi = lane >> 4;  // 0 or 1
  int tm = wid / tiles_n;
  int tn = wid - tm * tiles_n;

  int m0 = tm * 16 + hi * 8;   // C/D row base for this half-wave
  int n0 = tn * 16 + ln;       // output column

  float* Op = O + m0 * CD + n0;          // < 2^31, 32-bit indexing
  v8f c;
#pragma unroll
  for (int r = 0; r < 8; ++r) c[r] = Op[r * CD];

  const float* Arow = A + (tm * 16 + ln) * C_HID;
  const v16bf* Bcol = (const v16bf*)(Wt + n0 * C_HID);  // column-major bf16, 32B frags

#pragma unroll
  for (int kk = 0; kk < C_HID / 32; ++kk) {
    v16bf a;
    {
      int kb = kk * 32 + hi * 16;
      const float4* p0 = (const float4*)(Arow + kb);       // K = kb .. kb+7
      const float4* p1 = (const float4*)(Arow + kb + 8);   // K = kb+8 .. kb+15
      float4 l0 = p0[0], l1 = p0[1], h0 = p1[0], h1 = p1[1];
      a[0]  = (__bf16)l0.x; a[1]  = (__bf16)l0.y; a[2]  = (__bf16)l0.z; a[3]  = (__bf16)l0.w;
      a[4]  = (__bf16)l1.x; a[5]  = (__bf16)l1.y; a[6]  = (__bf16)l1.z; a[7]  = (__bf16)l1.w;
      a[8]  = (__bf16)h0.x; a[9]  = (__bf16)h0.y; a[10] = (__bf16)h0.z; a[11] = (__bf16)h0.w;
      a[12] = (__bf16)h1.x; a[13] = (__bf16)h1.y; a[14] = (__bf16)h1.z; a[15] = (__bf16)h1.w;
    }
    v16bf b = Bcol[kk * 2 + hi];  // one contiguous 32-byte fragment
    c = __builtin_amdgcn_wmma_f32_16x16x32_bf16(false, a, false, b, (short)0, c,
                                                false, false);
  }

#pragma unroll
  for (int r = 0; r < 8; ++r) Op[r * CD] = c[r];
}

// ---------------- BatchNorm (training-mode batch stats) ----------------

__global__ void k_bn_stats(const float* __restrict__ X, int n, int CD,
                           const float* __restrict__ g, float* __restrict__ stats) {
  int c = blockIdx.x;  // one block per channel
  float s = 0.0f, s2 = 0.0f;
  for (int i = threadIdx.x; i < n; i += blockDim.x) {
    float v = X[(size_t)i * CD + c];
    s += v;
    s2 += v * v;
  }
  __shared__ float sh[256];
  __shared__ float sh2[256];
  sh[threadIdx.x] = s;
  sh2[threadIdx.x] = s2;
  __syncthreads();
  for (int off = 128; off > 0; off >>= 1) {
    if (threadIdx.x < off) {
      sh[threadIdx.x] += sh[threadIdx.x + off];
      sh2[threadIdx.x] += sh2[threadIdx.x + off];
    }
    __syncthreads();
  }
  if (threadIdx.x == 0) {
    float mu = sh[0] / (float)n;
    float var = sh2[0] / (float)n - mu * mu;
    stats[c] = mu;
    stats[CD + c] = g[c] * rsqrtf(var + 1e-5f);  // gamma folded into scale
  }
}

__global__ void k_bn_apply(float* __restrict__ X, long total, int CD,
                           const float* __restrict__ stats,
                           const float* __restrict__ beta, int relu) {
  long i = (long)blockIdx.x * blockDim.x + threadIdx.x;
  long s = (long)gridDim.x * blockDim.x;
  for (; i < total; i += s) {
    int c = (int)(i & (CD - 1));  // CD is 64 or 128
    float v = (X[i] - stats[c]) * stats[CD + c] + beta[c];
    X[i] = relu ? fmaxf(v, 0.0f) : v;
  }
}

// ---------------- host orchestration ----------------

extern "C" void kernel_launch(void* const* d_in, const int* in_sizes, int n_in,
                              void* d_out, int out_size, void* d_ws, size_t ws_size,
                              hipStream_t stream) {
  (void)in_sizes; (void)n_in; (void)out_size; (void)ws_size;

  const float* x     = (const float*)d_in[0];
  const int*   ei    = (const int*)d_in[1];
  const float* W1    = (const float*)d_in[2];
  const float* g1    = (const float*)d_in[4];
  const float* beta1 = (const float*)d_in[5];
  const float* W2    = (const float*)d_in[6];
  const float* g2    = (const float*)d_in[8];
  const float* beta2 = (const float*)d_in[9];
  const float* W3    = (const float*)d_in[10];
  const float* g3    = (const float*)d_in[12];
  const float* beta3 = (const float*)d_in[13];

  float* out_mu = (float*)d_out;
  float* out_ls = out_mu + (size_t)N_NODES * C_OUT;

  const int* src = ei;
  const int* dst = ei + E_EDGES;

  float* ws = (float*)d_ws;
  float* deg   = ws; ws += N_NODES;
  float* dinv  = ws; ws += N_NODES;
  float* ew    = ws; ws += E_EDGES;
  float* stats = ws; ws += 2 * C_HID;
  __bf16* W1t = (__bf16*)ws; ws += (size_t)K1 * C_HID * C_HID / 2;  // bf16, counted in floats
  __bf16* W2t = (__bf16*)ws; ws += (size_t)K2 * C_OUT * C_HID / 2;
  __bf16* W3t = (__bf16*)ws; ws += (size_t)K2 * C_OUT * C_HID / 2;
  float* Ta = ws; ws += (size_t)N_NODES * C_HID;
  float* Tb = ws; ws += (size_t)N_NODES * C_HID;
  float* H  = ws; ws += (size_t)N_NODES * C_HID;

  const long NH = (long)N_NODES * C_HID;

  auto ZERO = [&](float* p, long n) {
    k_zero<<<2048, 256, 0, stream>>>(p, n);
  };
  auto INITNEG = [&](float* o, const float* a) {
    k_initneg<<<2048, 256, 0, stream>>>((float4*)o, (const float4*)a, NH / 4);
  };
  auto SPMM = [&](const float* Z, float* Pout, float scale) {
    long items = (long)E_EDGES * 32;
    k_spmm<<<(unsigned)((items + 255) / 256), 256, 0, stream>>>(Z, Pout, src, dst, ew,
                                                               scale, E_EDGES);
  };
  auto GEMM = [&](const float* A, const __bf16* Wtk, float* O, int CD) {
    int tiles_n = CD / 16;
    int total = (N_NODES / 16) * tiles_n;
    int blocks = (total + 7) / 8;  // 8 waves (256 threads) per block
    k_gemm_bf16<<<blocks, 256, 0, stream>>>(A, Wtk, O, CD, tiles_n, total);
  };

  // --- Laplacian edge weights: w = -dinv[src]*dinv[dst] ---
  ZERO(deg, N_NODES);
  k_deg<<<(E_EDGES + 255) / 256, 256, 0, stream>>>(src, deg, E_EDGES);
  k_dinv<<<(N_NODES + 255) / 256, 256, 0, stream>>>(deg, dinv, N_NODES);
  k_edgew<<<(E_EDGES + 255) / 256, 256, 0, stream>>>(src, dst, dinv, ew, E_EDGES);

  // --- Weights -> bf16, transposed for contiguous B fragments ---
  k_wt<<<1024, 256, 0, stream>>>(W1, W1t, K1, C_HID);
  k_wt<<<512, 256, 0, stream>>>(W2, W2t, K2, C_OUT);
  k_wt<<<512, 256, 0, stream>>>(W3, W3t, K2, C_OUT);

  // --- Layer 1: H = sum_k Tx_k @ W1[k] (bias cancels under BN) ---
  ZERO(H, NH);
  GEMM(x, W1t, H, C_HID);                               // k = 0, Tx0 = x
  ZERO(Ta, NH);
  SPMM(x, Ta, 1.0f);                                    // Ta = Tx1 = L_hat x
  GEMM(Ta, W1t + (size_t)1 * C_HID * C_HID, H, C_HID);
  {
    const float* prev = x;
    float* cur = Ta;
    for (int k = 2; k < K1; ++k) {
      float* nxt = (k == 2) ? Tb : (float*)prev;  // reuse Tx_{k-2}'s buffer (in-place neg ok)
      INITNEG(nxt, prev);                         // nxt = -Tx_{k-2}
      SPMM(cur, nxt, 2.0f);                       // nxt += 2*L_hat*Tx_{k-1}
      GEMM(nxt, W1t + (size_t)k * C_HID * C_HID, H, C_HID);
      prev = cur;
      cur = nxt;
    }
  }
  // BN + ReLU (in place): H becomes h
  k_bn_stats<<<C_HID, 256, 0, stream>>>(H, N_NODES, C_HID, g1, stats);
  k_bn_apply<<<2048, 256, 0, stream>>>(H, NH, C_HID, stats, beta1, 1);

  // --- Layers 2 & 3 share the Chebyshev basis Tx_k(h) ---
  ZERO(out_mu, (long)N_NODES * C_OUT);
  ZERO(out_ls, (long)N_NODES * C_OUT);
  GEMM(H, W2t, out_mu, C_OUT);                          // k = 0
  GEMM(H, W3t, out_ls, C_OUT);
  ZERO(Ta, NH);
  SPMM(H, Ta, 1.0f);                                    // Ta = Tx1(h)
  GEMM(Ta, W2t + (size_t)1 * C_OUT * C_HID, out_mu, C_OUT);
  GEMM(Ta, W3t + (size_t)1 * C_OUT * C_HID, out_ls, C_OUT);
  {
    const float* prev = H;
    float* cur = Ta;
    for (int k = 2; k < K2; ++k) {
      float* nxt = (k == 2) ? Tb : (float*)prev;
      INITNEG(nxt, prev);
      SPMM(cur, nxt, 2.0f);
      GEMM(nxt, W2t + (size_t)k * C_OUT * C_HID, out_mu, C_OUT);
      GEMM(nxt, W3t + (size_t)k * C_OUT * C_HID, out_ls, C_OUT);
      prev = cur;
      cur = nxt;
    }
  }
  // BN (no ReLU) on both heads
  k_bn_stats<<<C_OUT, 256, 0, stream>>>(out_mu, N_NODES, C_OUT, g2, stats);
  k_bn_apply<<<2048, 256, 0, stream>>>(out_mu, (long)N_NODES * C_OUT, C_OUT, stats, beta2, 0);
  k_bn_stats<<<C_OUT, 256, 0, stream>>>(out_ls, N_NODES, C_OUT, g3, stats);
  k_bn_apply<<<2048, 256, 0, stream>>>(out_ls, (long)N_NODES * C_OUT, C_OUT, stats, beta3, 0);
}
